// DrugEncoder_72335839199973
// MI455X (gfx1250) — matrix operand order repaired and checked by reading, not compiled, not run on hardware
//
#include <hip/hip_runtime.h>

#define NNODES  50000
#define EMB     768
#define INDIM   82
#define OUTDIM  128
#define NLAYERS 5
#define NEDGES  100000
#define BN_EPS  1e-5f

typedef __attribute__((ext_vector_type(16))) __bf16 v16bf;
typedef __attribute__((ext_vector_type(8)))  __bf16 v8bf;
typedef __attribute__((ext_vector_type(8)))  float  v8f;

#if defined(__gfx1250__) && __has_builtin(__builtin_amdgcn_global_load_async_to_lds_b128) && __has_builtin(__builtin_amdgcn_s_wait_asynccnt)
#define USE_ASYNC 1
typedef int v4i __attribute__((ext_vector_type(4)));
typedef __attribute__((address_space(1))) v4i v4i_glob;   // global (AS1)
typedef __attribute__((address_space(3))) v4i v4i_lds;    // LDS (AS3)
#else
#define USE_ASYNC 0
#endif

__device__ __forceinline__ void wait_async_lds() {
#if USE_ASYNC
    __builtin_amdgcn_s_wait_asynccnt(0);
#endif
}

__device__ __forceinline__ void copy16B_to_lds(const void* gp, void* lp) {
#if USE_ASYNC
    __builtin_amdgcn_global_load_async_to_lds_b128((v4i_glob*)gp, (v4i_lds*)lp, 0, 0);
#else
    *(float4*)lp = *(const float4*)gp;
#endif
}

// ---------------------------------------------------------------------------
// Weight prep: W [K,N] fp32 row-major  ->  Wt [N,K] bf16 row-major
// ---------------------------------------------------------------------------
__global__ void wt_transpose_bf16(const float* __restrict__ W, __bf16* __restrict__ Wt,
                                  int K, int N) {
    int idx = blockIdx.x * blockDim.x + threadIdx.x;
    if (idx < K * N) {
        int k = idx / N;
        int n = idx % N;
        Wt[(long)n * K + k] = (__bf16)W[idx];
    }
}

// ---------------------------------------------------------------------------
// GEMM: C[M,N] = A[M,K](f32) x Bt[N,K](bf16) (+ bias), f32 accumulate.
// 256 threads = 8 wave32; 128x128 tile; BLOCK_K = 32; v_wmma_f32_16x16x32_bf16.
// A staged to LDS as f32 (pure copy -> GLOBAL_LOAD_ASYNC_TO_LDS_B128),
// converted to bf16 at fragment load. Ping-pong LDS overlaps staging w/ WMMA.
// FULL_K=true: K%32==0 and rows 16B-aligned -> fast path only (no edge code).
// ---------------------------------------------------------------------------
template <bool FULL_K>
__launch_bounds__(256)
__global__ void gemm_bf16_wmma(const float* __restrict__ A, const __bf16* __restrict__ Bt,
                               float* __restrict__ C, const float* __restrict__ bias,
                               int M, int N, int K) {
    __shared__ __align__(16) float  AsF[2][128 * 36];  // M-major, K padded 32->36 f32
    __shared__ __align__(16) __bf16 Bs [2][128 * 40];  // N-major, K padded 32->40 bf16

    const int tid  = threadIdx.x;
    const int lane = tid & 31;           // wave32
    const int wid  = tid >> 5;           // 0..7
    const int wm   = wid & 3;            // wave row group: 32 rows each
    const int wn   = wid >> 2;           // wave col group: 64 cols each
    const int bm   = blockIdx.x * 128;
    const int bn   = blockIdx.y * 128;

    v8f acc[2][4];
#pragma unroll
    for (int i = 0; i < 2; ++i)
#pragma unroll
        for (int j = 0; j < 4; ++j) acc[i][j] = v8f{};

    const int srow = tid >> 1;           // 0..127
    const int scol = (tid & 1) * 16;     // element offset 0 or 16 within 32-wide K chunk

    // ---- staging helpers --------------------------------------------------
    auto stageA = [&](int buf, int k0) {
        const int gr = bm + srow;
        float* dstp = &AsF[buf][srow * 36 + scol];
        if (FULL_K) {
            // K%32==0 and K%4==0 guaranteed; only row guard needed.
            // OOB rows leave stale LDS: row m of C depends only on row m of A,
            // and those rows are never stored in the epilogue -> harmless.
            if (gr < M) {
                const float* gp = A + (long)gr * K + k0 + scol;
#pragma unroll
                for (int q = 0; q < 4; ++q)
                    copy16B_to_lds(gp + q * 4, dstp + q * 4);
            }
        } else if (gr < M && (k0 + 32) <= K && (K & 3) == 0) {
            const float* gp = A + (long)gr * K + k0 + scol;
#pragma unroll
            for (int q = 0; q < 4; ++q)
                copy16B_to_lds(gp + q * 4, dstp + q * 4);
        } else {
            const int grc = gr < M ? gr : M - 1;
#pragma unroll
            for (int j = 0; j < 16; ++j) {
                int k  = k0 + scol + j;
                int kc = k < K ? k : K - 1;
                float v = A[(long)grc * K + kc];      // clamped: always in-bounds
                if (gr >= M || k >= K) v = 0.0f;
                dstp[j] = v;
            }
        }
    };
    auto stageB = [&](int buf, int k0) {
        const int gn = bn + srow;                     // N multiple of 128: always < N
        __bf16* dstp = &Bs[buf][srow * 40 + scol];
        if (FULL_K || ((k0 + 32) <= K && (K & 7) == 0)) {
            const __bf16* gp = Bt + (long)gn * K + k0 + scol;  // 16B aligned
            copy16B_to_lds(gp, dstp);
            copy16B_to_lds(gp + 8, dstp + 8);
        } else {
#pragma unroll
            for (int j = 0; j < 16; ++j) {
                int k  = k0 + scol + j;
                int kc = k < K ? k : 0;
                __bf16 v = Bt[(long)gn * K + kc];
                if (k >= K) v = (__bf16)0.0f;
                dstp[j] = v;
            }
        }
    };

    // ---- compute: fragments in contiguous k order (k = 16*half + s), f32->bf16 convert for A --
    auto computeTile = [&](int buf) {
        const int rl = lane & 15;
        const int ah = ((lane >> 4) & 1) * 16;   // A: lanes<16 -> K 0-15, else K 16-31
        const int bh = ((lane >> 4) & 1) * 16;   // B: lanes<16 -> K 0-15, else K 16-31
        v16bf afrag[2], bfrag[4];
#pragma unroll
        for (int mi = 0; mi < 2; ++mi) {
            const float* ab = &AsF[buf][(wm * 32 + mi * 16 + rl) * 36];
#pragma unroll
            for (int j = 0; j < 8; ++j) {
                afrag[mi][j]     = (__bf16)ab[ah + j];
                afrag[mi][j + 8] = (__bf16)ab[ah + 8 + j];
            }
        }
#pragma unroll
        for (int ni = 0; ni < 4; ++ni) {
            const __bf16* bb = &Bs[buf][(wn * 64 + ni * 16 + rl) * 40];
            union { v16bf v; v8bf h[2]; } u;
            u.h[0] = *(const v8bf*)(bb + bh);
            u.h[1] = *(const v8bf*)(bb + bh + 8);
            bfrag[ni] = u.v;
        }
#pragma unroll
        for (int mi = 0; mi < 2; ++mi)
#pragma unroll
            for (int ni = 0; ni < 4; ++ni)
                acc[mi][ni] = __builtin_amdgcn_wmma_f32_16x16x32_bf16(
                    false, afrag[mi], false, bfrag[ni], (short)0, acc[mi][ni], false, false);
    };

    // ---- ping-pong pipeline ----------------------------------------------
    const int nch = (K + 31) / 32;
    stageA(0, 0);
    stageB(0, 0);
    wait_async_lds();
    __syncthreads();
    for (int c = 0; c < nch; ++c) {
        const int cur = c & 1;
        if (c + 1 < nch) {                 // prefetch next tile into other buffer
            stageA(cur ^ 1, (c + 1) * 32);
            stageB(cur ^ 1, (c + 1) * 32);
        }
        computeTile(cur);
        wait_async_lds();
        __syncthreads();
    }

    // ---- epilogue: C/D layout = VGPR j -> M=j (lanes 0-15) / M=j+8 (16-31) ----
    const int cl  = lane & 15;
    const int rhi = ((lane >> 4) & 1) * 8;
#pragma unroll
    for (int mi = 0; mi < 2; ++mi) {
#pragma unroll
        for (int ni = 0; ni < 4; ++ni) {
            const int gc = bn + wn * 64 + ni * 16 + cl;
            const float bv = bias ? bias[gc] : 0.0f;
#pragma unroll
            for (int j = 0; j < 8; ++j) {
                const int gr = bm + wm * 32 + mi * 16 + rhi + j;
                if (gr < M) C[(long)gr * N + gc] = acc[mi][ni][j] + bv;
            }
        }
    }
}

// ---------------------------------------------------------------------------
// Degree / normalization
// ---------------------------------------------------------------------------
__global__ void deg_init_kernel(float* deg, int n) {
    int i = blockIdx.x * blockDim.x + threadIdx.x;
    if (i < n) deg[i] = 1.0f;  // self loop
}
__global__ void deg_edge_kernel(const int* __restrict__ dst, float* deg, int e) {
    int i = blockIdx.x * blockDim.x + threadIdx.x;
    if (i < e) unsafeAtomicAdd(&deg[dst[i]], 1.0f);
}
__global__ void dinv_kernel(const float* __restrict__ deg, float* dinv, int n) {
    int i = blockIdx.x * blockDim.x + threadIdx.x;
    if (i < n) dinv[i] = rsqrtf(deg[i]);
}

// ---------------------------------------------------------------------------
// GCN aggregation: out[v] = t[v]*dinv[v]^2 + bias  (init, writes everything)
//                  out[dst] += t[src]*dinv[src]*dinv[dst]  (edge atomics)
// ---------------------------------------------------------------------------
__global__ void agg_init_kernel(const float* __restrict__ t, const float* __restrict__ dinv,
                                const float* __restrict__ bias, float* __restrict__ out,
                                int n, int F) {
    int v = blockIdx.x;
    int c = blockIdx.y * blockDim.x + threadIdx.x;
    if (c < F) {
        float di = dinv[v];
        out[(long)v * F + c] = t[(long)v * F + c] * di * di + (bias ? bias[c] : 0.0f);
    }
}
__global__ void agg_edge_kernel(const float* __restrict__ t, const float* __restrict__ dinv,
                                const int* __restrict__ src, const int* __restrict__ dst,
                                float* __restrict__ out, int E, int F) {
    int e = blockIdx.x;
    int c = blockIdx.y * blockDim.x + threadIdx.x;
    if (c < F) {
        int s = src[e], d = dst[e];
        float w = dinv[s] * dinv[d];
        unsafeAtomicAdd(&out[(long)d * F + c], t[(long)s * F + c] * w);
    }
}

// ---------------------------------------------------------------------------
// BatchNorm (two-pass, biased variance like the reference) + ReLU + residual
// ---------------------------------------------------------------------------
__global__ void zero_kernel(float* p, int n) {
    int i = blockIdx.x * blockDim.x + threadIdx.x;
    if (i < n) p[i] = 0.0f;
}
__global__ void bn_stats_kernel(const float* __restrict__ x, float* sum, float* sumsq,
                                int n, int F) {
    int c = blockIdx.y * blockDim.x + threadIdx.x;
    if (c >= F) return;
    float s = 0.0f, ss = 0.0f;
    for (int r = blockIdx.x; r < n; r += gridDim.x) {
        float v = x[(long)r * F + c];
        s += v; ss += v * v;
    }
    unsafeAtomicAdd(&sum[c], s);
    unsafeAtomicAdd(&sumsq[c], ss);
}
__global__ void bn_finalize_kernel(const float* sum, const float* sumsq,
                                   const float* __restrict__ gamma, const float* __restrict__ beta,
                                   float* scale, float* shift, int n, int F) {
    int c = blockIdx.x * blockDim.x + threadIdx.x;
    if (c >= F) return;
    float inv_n = 1.0f / (float)n;
    float mu  = sum[c] * inv_n;
    float var = sumsq[c] * inv_n - mu * mu;
    float sc  = gamma[c] * rsqrtf(var + BN_EPS);
    scale[c] = sc;
    shift[c] = beta[c] - mu * sc;
}
// vectorized: h += relu(hn*scale+shift), 4 floats/thread (F % 4 == 0)
__global__ void bn_relu_res_kernel(float4* __restrict__ h4, const float4* __restrict__ hn4,
                                   const float* __restrict__ scale, const float* __restrict__ shift,
                                   long total4, int F4) {
    long i = (long)blockIdx.x * blockDim.x + threadIdx.x;
    if (i < total4) {
        int c = (int)(i % F4) * 4;
        float4 v = hn4[i];
        float4 r = h4[i];
        float t0 = v.x * scale[c + 0] + shift[c + 0];
        float t1 = v.y * scale[c + 1] + shift[c + 1];
        float t2 = v.z * scale[c + 2] + shift[c + 2];
        float t3 = v.w * scale[c + 3] + shift[c + 3];
        r.x += t0 > 0.0f ? t0 : 0.0f;
        r.y += t1 > 0.0f ? t1 : 0.0f;
        r.z += t2 > 0.0f ? t2 : 0.0f;
        r.w += t3 > 0.0f ? t3 : 0.0f;
        h4[i] = r;
    }
}

// ---------------------------------------------------------------------------
extern "C" void kernel_launch(void* const* d_in, const int* in_sizes, int n_in,
                              void* d_out, int out_size, void* d_ws, size_t ws_size,
                              hipStream_t stream) {
    const float* x      = (const float*)d_in[0];
    const float* W_in   = (const float*)d_in[1];
    const float* b_in   = (const float*)d_in[2];
    const float* Wc     = (const float*)d_in[3];
    const float* bc     = (const float*)d_in[4];
    const float* gamma  = (const float*)d_in[5];
    const float* beta   = (const float*)d_in[6];
    const float* W_proj = (const float*)d_in[7];
    const float* b_proj = (const float*)d_in[8];
    const float* W_out  = (const float*)d_in[9];
    const float* b_out  = (const float*)d_in[10];
    const int*   ei     = (const int*)d_in[11];
    const int*   src    = ei;
    const int*   dst    = ei + NEDGES;
    float*       outp   = (float*)d_out;

    // workspace carve-up (256B aligned regions)
    char* w = (char*)d_ws;
    auto carve = [&](size_t bytes) { char* p = w; w += (bytes + 255) & ~(size_t)255; return p; };
    float*  h     = (float*)carve(sizeof(float) * (size_t)NNODES * EMB);
    float*  t     = (float*)carve(sizeof(float) * (size_t)NNODES * EMB);
    float*  hn    = (float*)carve(sizeof(float) * (size_t)NNODES * EMB);
    float*  deg   = (float*)carve(sizeof(float) * NNODES);
    float*  dinv  = (float*)carve(sizeof(float) * NNODES);
    float*  bnsum = (float*)carve(sizeof(float) * EMB);
    float*  bnsq  = (float*)carve(sizeof(float) * EMB);
    float*  scale = (float*)carve(sizeof(float) * EMB);
    float*  shift = (float*)carve(sizeof(float) * EMB);
    __bf16* WtIn  = (__bf16*)carve(sizeof(__bf16) * (size_t)EMB * INDIM);
    __bf16* WtC   = (__bf16*)carve(sizeof(__bf16) * (size_t)NLAYERS * EMB * EMB);
    __bf16* WtP   = (__bf16*)carve(sizeof(__bf16) * (size_t)EMB * EMB);
    __bf16* WtO   = (__bf16*)carve(sizeof(__bf16) * (size_t)OUTDIM * EMB);

    // --- weight prep (bf16, transposed to [N,K]) ---
    wt_transpose_bf16<<<(INDIM * EMB + 255) / 256, 256, 0, stream>>>(W_in, WtIn, INDIM, EMB);
    for (int l = 0; l < NLAYERS; ++l)
        wt_transpose_bf16<<<(EMB * EMB + 255) / 256, 256, 0, stream>>>(
            Wc + (size_t)l * EMB * EMB, WtC + (size_t)l * EMB * EMB, EMB, EMB);
    wt_transpose_bf16<<<(EMB * EMB + 255) / 256, 256, 0, stream>>>(W_proj, WtP, EMB, EMB);
    wt_transpose_bf16<<<(EMB * OUTDIM + 255) / 256, 256, 0, stream>>>(W_out, WtO, EMB, OUTDIM);

    // --- degrees / dinv ---
    deg_init_kernel<<<(NNODES + 255) / 256, 256, 0, stream>>>(deg, NNODES);
    deg_edge_kernel<<<(NEDGES + 255) / 256, 256, 0, stream>>>(dst, deg, NEDGES);
    dinv_kernel<<<(NNODES + 255) / 256, 256, 0, stream>>>(deg, dinv, NNODES);

    const dim3 gemm_blk(256);
    const int  mtiles = (NNODES + 127) / 128;
    const long total4 = (long)NNODES * EMB / 4;

    // --- input projection: h = x @ W_in + b_in  (K=82 -> generic path) ---
    gemm_bf16_wmma<false><<<dim3(mtiles, EMB / 128), gemm_blk, 0, stream>>>(
        x, WtIn, h, b_in, NNODES, EMB, INDIM);

    // --- 5 GCN layers (K=768 -> fast path only) ---
    for (int l = 0; l < NLAYERS; ++l) {
        gemm_bf16_wmma<true><<<dim3(mtiles, EMB / 128), gemm_blk, 0, stream>>>(
            h, WtC + (size_t)l * EMB * EMB, t, nullptr, NNODES, EMB, EMB);
        agg_init_kernel<<<dim3(NNODES, (EMB + 255) / 256), 256, 0, stream>>>(
            t, dinv, bc + (size_t)l * EMB, hn, NNODES, EMB);
        agg_edge_kernel<<<dim3(NEDGES, (EMB + 255) / 256), 256, 0, stream>>>(
            t, dinv, src, dst, hn, NEDGES, EMB);
        zero_kernel<<<(EMB + 255) / 256, 256, 0, stream>>>(bnsum, EMB);
        zero_kernel<<<(EMB + 255) / 256, 256, 0, stream>>>(bnsq, EMB);
        bn_stats_kernel<<<dim3(64, (EMB + 255) / 256), 256, 0, stream>>>(hn, bnsum, bnsq, NNODES, EMB);
        bn_finalize_kernel<<<(EMB + 255) / 256, 256, 0, stream>>>(
            bnsum, bnsq, gamma + (size_t)l * EMB, beta + (size_t)l * EMB, scale, shift, NNODES, EMB);
        bn_relu_res_kernel<<<(int)((total4 + 255) / 256), 256, 0, stream>>>(
            (float4*)h, (const float4*)hn, scale, shift, total4, EMB / 4);
    }

    // --- output projection: t = h @ W_proj + b_proj ---
    gemm_bf16_wmma<true><<<dim3(mtiles, EMB / 128), gemm_blk, 0, stream>>>(
        h, WtP, t, b_proj, NNODES, EMB, EMB);

    // --- final GCNConv: g = t @ W_out (into hn), then aggregate into d_out ---
    gemm_bf16_wmma<true><<<dim3(mtiles, OUTDIM / 128), gemm_blk, 0, stream>>>(
        t, WtO, hn, nullptr, NNODES, OUTDIM, EMB);
    agg_init_kernel<<<dim3(NNODES, (OUTDIM + 255) / 256), 256, 0, stream>>>(
        hn, dinv, b_out, outp, NNODES, OUTDIM);
    agg_edge_kernel<<<dim3(NEDGES, (OUTDIM + 255) / 256), 256, 0, stream>>>(
        hn, dinv, src, dst, outp, NEDGES, OUTDIM);
}
